// ScaledDotProductAttention_65309272703203
// MI455X (gfx1250) — matrix operand order repaired and checked by reading, not compiled, not run on hardware
//
#include <hip/hip_runtime.h>
#include <hip/hip_bf16.h>

// ---------------------------------------------------------------------------
// Multi-head attention (b=2, n=2048, d_model=1024, H=16, d_k=64) on gfx1250.
// Matmuls via v_wmma_f32_16x16x32_bf16; shared B tiles staged to LDS with
// global_load_async_to_lds_b128 (ASYNCcnt) and double buffering.
// Per-wave tile: 32(M) x 64(N), 8 accumulators, batched fragment loads.
// ---------------------------------------------------------------------------

typedef __attribute__((ext_vector_type(16))) __bf16 v16bf;
typedef __attribute__((ext_vector_type(8)))  __bf16 v8bf;
typedef __attribute__((ext_vector_type(4)))  __bf16 v4bf;
typedef __attribute__((ext_vector_type(8)))  float  v8f;
typedef __attribute__((ext_vector_type(4)))  float  v4f;

#define B_SZ   2
#define NSEQ   2048
#define DMODEL 1024
#define NHEAD  16
#define DK     64
#define MROWS  (B_SZ * NSEQ)             // 4096
#define OUT0_ELEMS (2ull * 2048 * 1024)  // out_final elements

#define KC     128                       // K-chunk staged per LDS buffer
#define KCPAD  136                       // +8 bf16 pad: 272B row stride, no bank conflicts

// ----- WMMA helpers --------------------------------------------------------

__device__ __forceinline__ v8f wmma_bf16(v16bf a, v16bf b, v8f c) {
  // 8 args: (neg_a, A, neg_b, B, c_mod, C, reuse_a, reuse_b)
  return __builtin_amdgcn_wmma_f32_16x16x32_bf16(false, a, false, b,
                                                 (short)0, c, false, false);
}

// A-matrix 16x32 bf16 fragment for one lane.
// lane<16 : elems 0..15 = K 0..15
// lane>=16: elems 0..15 = K 16..31
__device__ __forceinline__ v16bf load_a_bf16(const __bf16* rowK0, int half) {
  v8bf lo = *(const v8bf*)(rowK0 + 16 * half);
  v8bf hi = *(const v8bf*)(rowK0 + 16 * half + 8);
  v16bf a;
#pragma unroll
  for (int i = 0; i < 8; ++i) { a[i] = lo[i]; a[i + 8] = hi[i]; }
  return a;
}

// Same fragment, converting from f32 source on the fly.
__device__ __forceinline__ v16bf cvt_a_f32(const float* rowK0, int half) {
  const float* lo = rowK0 + 16 * half;
  const float* hi = rowK0 + 16 * half + 8;
  v16bf a;
#pragma unroll
  for (int i = 0; i < 8; ++i) { a[i] = (__bf16)lo[i]; a[i + 8] = (__bf16)hi[i]; }
  return a;
}

// B-matrix 32x16 bf16 fragment: lane holds 16 contiguous K values of column n,
// lanes 0-15 -> K 0..15, lanes 16-31 -> K 16..31.
__device__ __forceinline__ v16bf load_b_bf16(const __bf16* colK0, int half) {
  return *(const v16bf*)(colK0 + 16 * half);
}

// ----- async global -> LDS (CDNA5, ASYNCcnt) -------------------------------

__device__ __forceinline__ void async_ld16(unsigned lds_off, unsigned long long gaddr) {
  asm volatile("global_load_async_to_lds_b128 %0, %1, off"
               :: "v"(lds_off), "v"(gaddr) : "memory");
}

__device__ __forceinline__ void wait_async0() {
#if __has_builtin(__builtin_amdgcn_s_wait_asynccnt)
  __builtin_amdgcn_s_wait_asynccnt(0);
#else
  asm volatile("s_wait_asynccnt 0x0" ::: "memory");
#endif
}

// Stage a 64-row x 128-K bf16 tile of row-major B (row stride LDB elements)
// into LDS. 256 threads x 4 chunks x 16B = 16 KB.
template <int LDB>
__device__ __forceinline__ void stage_b_tile(const __bf16* __restrict__ B,
                                             int row0, int k0,
                                             __bf16 (*tile)[KCPAD], int tid) {
#pragma unroll
  for (int j = 0; j < 4; ++j) {
    int c   = tid + j * 256;      // chunk id 0..1023
    int row = c >> 4;             // 0..63
    int kc  = (c & 15) * 8;       // element offset within chunk row
    unsigned lds = (unsigned)(size_t)&tile[row][kc];
    const __bf16* g = B + (size_t)(row0 + row) * LDB + k0 + kc;
    async_ld16(lds, (unsigned long long)(size_t)g);
  }
}

// ----- elementwise f32 -> bf16 --------------------------------------------

__global__ void f32_to_bf16_kernel(const float* __restrict__ in,
                                   __bf16* __restrict__ out, size_t n) {
  size_t i = ((size_t)blockIdx.x * blockDim.x + threadIdx.x) * 4;
  if (i + 3 < n) {
    v4f v = *(const v4f*)(in + i);
    v4bf o;
#pragma unroll
    for (int j = 0; j < 4; ++j) o[j] = (__bf16)v[j];
    *(v4bf*)(out + i) = o;
  }
}

// ----- NT GEMM: C[m,n] = sum_k X[m,k]*W[n,k] + bias[n]  --------------------
// modes: 0=Q proj (mask=query_mask, out [bh][q][dk] bf16)
//        1=K proj (mask=key_mask,   out [bh][k][dk] bf16)
//        2=V proj (mask=key_mask,   out transposed [bh][dk][k] bf16)
//        3=output proj (no mask, out f32 row-major [row][n])
__global__ __launch_bounds__(256) void gemm_nt_bf16(
    const __bf16* __restrict__ X, const __bf16* __restrict__ W,
    const float* __restrict__ bias, const float* __restrict__ tokmask,
    __bf16* __restrict__ outb, float* __restrict__ outf, int mode) {
  __shared__ __bf16 tileB[2][64][KCPAD];

  const int tid  = threadIdx.x;
  const int wave = tid >> 5;
  const int lane = tid & 31;
  const int half = lane >> 4;
  const int nl   = lane & 15;

  const int row_base = blockIdx.x * 256 + wave * 32;  // M tile (32 rows/wave)
  const int col_base = blockIdx.y * 64;               // N tile (64 cols/block)

  v8f acc[8] = {};
  const __bf16* arow0 = X + (size_t)(row_base + nl) * DMODEL;
  const __bf16* arow1 = arow0 + (size_t)16 * DMODEL;

  stage_b_tile<DMODEL>(W, col_base, 0, tileB[0], tid);
  int buf = 0;
  for (int kc = 0; kc < DMODEL; kc += KC) {
    wait_async0();
    __syncthreads();
    if (kc + KC < DMODEL)
      stage_b_tile<DMODEL>(W, col_base, kc + KC, tileB[buf ^ 1], tid);
#pragma unroll
    for (int ks = 0; ks < KC; ks += 32) {
      v16bf a0 = load_a_bf16(arow0 + kc + ks, half);
      v16bf a1 = load_a_bf16(arow1 + kc + ks, half);
      v16bf b0 = *(const v16bf*)&tileB[buf][ 0 + nl][ks + 16 * half];
      v16bf b1 = *(const v16bf*)&tileB[buf][16 + nl][ks + 16 * half];
      v16bf b2 = *(const v16bf*)&tileB[buf][32 + nl][ks + 16 * half];
      v16bf b3 = *(const v16bf*)&tileB[buf][48 + nl][ks + 16 * half];
      acc[0] = wmma_bf16(a0, b0, acc[0]);
      acc[1] = wmma_bf16(a0, b1, acc[1]);
      acc[2] = wmma_bf16(a0, b2, acc[2]);
      acc[3] = wmma_bf16(a0, b3, acc[3]);
      acc[4] = wmma_bf16(a1, b0, acc[4]);
      acc[5] = wmma_bf16(a1, b1, acc[5]);
      acc[6] = wmma_bf16(a1, b2, acc[6]);
      acc[7] = wmma_bf16(a1, b3, acc[7]);
    }
    buf ^= 1;
  }

#pragma unroll
  for (int mh = 0; mh < 2; ++mh) {
#pragma unroll
    for (int t = 0; t < 4; ++t) {
#pragma unroll
      for (int r = 0; r < 8; ++r) {
        int row = row_base + mh * 16 + r + 8 * half;  // global M row
        int n   = col_base + t * 16 + nl;             // global N col
        int bi  = row >> 11;                          // batch
        int q   = row & (NSEQ - 1);                   // token
        float v = acc[mh * 4 + t][r] + bias[n];
        if (mode <= 2) {
          v *= tokmask[(bi << 11) + q];
          int h = n >> 6, d = n & 63;
          size_t bh = (size_t)bi * NHEAD + h;
          if (mode == 2)
            outb[(bh * DK + d) * NSEQ + q] = (__bf16)v;    // V transposed
          else
            outb[(bh * NSEQ + q) * DK + d] = (__bf16)v;    // Q / K
        } else {
          outf[(size_t)row * DMODEL + n] = v;              // final output
        }
      }
    }
  }
}

// ----- scores: S[bh,q,k] = (Q.K)/8 * attw, masked -> -inf ------------------

__global__ __launch_bounds__(256) void attn_scores_kernel(
    const __bf16* __restrict__ Qb, const __bf16* __restrict__ Kb,
    const float* __restrict__ attw, const unsigned char* __restrict__ amask,
    float* __restrict__ att) {
  const int tid  = threadIdx.x;
  const int wave = tid >> 5;
  const int lane = tid & 31;
  const int half = lane >> 4;
  const int nl   = lane & 15;

  const int bh = blockIdx.z;
  const int bi = bh >> 4;
  const int qb = blockIdx.y * 16;
  const int kb = blockIdx.x * 512 + wave * 64;

  v8f acc[4] = {};
  const __bf16* arow = Qb + ((size_t)bh * NSEQ + qb + nl) * DK;

#pragma unroll
  for (int kk = 0; kk < DK; kk += 32) {
    v16bf a  = load_a_bf16(arow + kk, half);
    v16bf b0 = load_b_bf16(Kb + ((size_t)bh * NSEQ + kb +  0 + nl) * DK + kk, half);
    v16bf b1 = load_b_bf16(Kb + ((size_t)bh * NSEQ + kb + 16 + nl) * DK + kk, half);
    v16bf b2 = load_b_bf16(Kb + ((size_t)bh * NSEQ + kb + 32 + nl) * DK + kk, half);
    v16bf b3 = load_b_bf16(Kb + ((size_t)bh * NSEQ + kb + 48 + nl) * DK + kk, half);
    acc[0] = wmma_bf16(a, b0, acc[0]);
    acc[1] = wmma_bf16(a, b1, acc[1]);
    acc[2] = wmma_bf16(a, b2, acc[2]);
    acc[3] = wmma_bf16(a, b3, acc[3]);
  }

#pragma unroll
  for (int t = 0; t < 4; ++t) {
#pragma unroll
    for (int r = 0; r < 8; ++r) {
      int q = qb + r + 8 * half;
      int k = kb + t * 16 + nl;
      size_t widx = ((size_t)bi * NSEQ + q) * NSEQ + k;
      float s = acc[t][r] * 0.125f * attw[widx];
      if (amask[widx]) s = -__builtin_inff();
      att[((size_t)bh * NSEQ + q) * NSEQ + k] = s;
    }
  }
}

// ----- row softmax over nk=2048, in place ----------------------------------

__global__ __launch_bounds__(256) void softmax_rows_kernel(float* __restrict__ att) {
  __shared__ float red[8];
  const int tid = threadIdx.x;
  float* p = att + (size_t)blockIdx.x * NSEQ;

  float x[8];
  float m = -__builtin_inff();
#pragma unroll
  for (int i = 0; i < 8; ++i) { x[i] = p[tid + i * 256]; m = fmaxf(m, x[i]); }
#pragma unroll
  for (int off = 16; off > 0; off >>= 1) m = fmaxf(m, __shfl_xor(m, off, 32));
  if ((tid & 31) == 0) red[tid >> 5] = m;
  __syncthreads();
  float bm = -__builtin_inff();
#pragma unroll
  for (int i = 0; i < 8; ++i) bm = fmaxf(bm, red[i]);
  __syncthreads();

  float s = 0.f;
#pragma unroll
  for (int i = 0; i < 8; ++i) { x[i] = __expf(x[i] - bm); s += x[i]; }
#pragma unroll
  for (int off = 16; off > 0; off >>= 1) s += __shfl_xor(s, off, 32);
  if ((tid & 31) == 0) red[tid >> 5] = s;
  __syncthreads();
  float bs = 0.f;
#pragma unroll
  for (int i = 0; i < 8; ++i) bs += red[i];
  float inv = 1.0f / bs;
#pragma unroll
  for (int i = 0; i < 8; ++i) p[tid + i * 256] = x[i] * inv;
}

// ----- PV: ctx[b,q,h*64+d] = sum_k P[bh,q,k] * V[bh,k,d] -------------------

__global__ __launch_bounds__(256) void attn_pv_kernel(
    const float* __restrict__ att, const __bf16* __restrict__ Vt,
    __bf16* __restrict__ ctx) {
  __shared__ __bf16 tileB[2][64][KCPAD];

  const int tid  = threadIdx.x;
  const int wave = tid >> 5;
  const int lane = tid & 31;
  const int half = lane >> 4;
  const int nl   = lane & 15;

  const int bh = blockIdx.y;
  const int bi = bh >> 4, h = bh & 15;
  const int qb = blockIdx.x * 256 + wave * 32;   // 32 q-rows per wave

  v8f acc[8] = {};
  const float*  arow0 = att + ((size_t)bh * NSEQ + qb + nl) * NSEQ;
  const float*  arow1 = arow0 + (size_t)16 * NSEQ;
  const __bf16* Vbase = Vt + (size_t)bh * DK * NSEQ;  // 64 rows (d), stride NSEQ

  stage_b_tile<NSEQ>(Vbase, 0, 0, tileB[0], tid);
  int buf = 0;
  for (int kc = 0; kc < NSEQ; kc += KC) {
    wait_async0();
    __syncthreads();
    if (kc + KC < NSEQ)
      stage_b_tile<NSEQ>(Vbase, 0, kc + KC, tileB[buf ^ 1], tid);
#pragma unroll
    for (int ks = 0; ks < KC; ks += 32) {
      v16bf a0 = cvt_a_f32(arow0 + kc + ks, half);
      v16bf a1 = cvt_a_f32(arow1 + kc + ks, half);
      v16bf b0 = *(const v16bf*)&tileB[buf][ 0 + nl][ks + 16 * half];
      v16bf b1 = *(const v16bf*)&tileB[buf][16 + nl][ks + 16 * half];
      v16bf b2 = *(const v16bf*)&tileB[buf][32 + nl][ks + 16 * half];
      v16bf b3 = *(const v16bf*)&tileB[buf][48 + nl][ks + 16 * half];
      acc[0] = wmma_bf16(a0, b0, acc[0]);
      acc[1] = wmma_bf16(a0, b1, acc[1]);
      acc[2] = wmma_bf16(a0, b2, acc[2]);
      acc[3] = wmma_bf16(a0, b3, acc[3]);
      acc[4] = wmma_bf16(a1, b0, acc[4]);
      acc[5] = wmma_bf16(a1, b1, acc[5]);
      acc[6] = wmma_bf16(a1, b2, acc[6]);
      acc[7] = wmma_bf16(a1, b3, acc[7]);
    }
    buf ^= 1;
  }

#pragma unroll
  for (int mh = 0; mh < 2; ++mh) {
#pragma unroll
    for (int t = 0; t < 4; ++t) {
#pragma unroll
      for (int r = 0; r < 8; ++r) {
        int q = qb + mh * 16 + r + 8 * half;
        int d = t * 16 + nl;
        ctx[((size_t)(bi * NSEQ + q)) * DMODEL + h * DK + d] =
            (__bf16)acc[mh * 4 + t][r];
      }
    }
  }
}

// ---------------------------------------------------------------------------

extern "C" void kernel_launch(void* const* d_in, const int* in_sizes, int n_in,
                              void* d_out, int out_size, void* d_ws, size_t ws_size,
                              hipStream_t stream) {
  const float* queries = (const float*)d_in[0];
  const float* keys    = (const float*)d_in[1];
  const float* values  = (const float*)d_in[2];
  const float* qmask   = (const float*)d_in[3];
  const float* kmask   = (const float*)d_in[4];
  const unsigned char* amask = (const unsigned char*)d_in[5];
  const float* attw    = (const float*)d_in[6];
  const float* Wq = (const float*)d_in[7];
  const float* Wk = (const float*)d_in[8];
  const float* Wv = (const float*)d_in[9];
  const float* Wo = (const float*)d_in[10];
  const float* bq = (const float*)d_in[11];
  const float* bk = (const float*)d_in[12];
  const float* bv = (const float*)d_in[13];
  const float* bo = (const float*)d_in[14];

  float* outF = (float*)d_out;              // out_final: 4096x1024 f32
  float* att  = outF + OUT0_ELEMS;          // att_softmax: [32][2048][2048] f32

  // workspace carve-out
  char* ws = (char*)d_ws;
  size_t off = 0;
  auto alloc = [&](size_t bytes) -> char* {
    char* p = ws + off;
    off += (bytes + 255) & ~(size_t)255;
    return p;
  };
  const size_t actB = (size_t)MROWS * DMODEL * sizeof(__bf16);     // 8 MB
  const size_t wB   = (size_t)DMODEL * DMODEL * sizeof(__bf16);    // 2 MB
  __bf16* Xq  = (__bf16*)alloc(actB);
  __bf16* Xk  = (__bf16*)alloc(actB);
  __bf16* Xv  = (__bf16*)alloc(actB);
  __bf16* Wqb = (__bf16*)alloc(wB);
  __bf16* Wkb = (__bf16*)alloc(wB);
  __bf16* Wvb = (__bf16*)alloc(wB);
  __bf16* Wob = (__bf16*)alloc(wB);
  __bf16* Qb  = (__bf16*)alloc(actB);   // [bh][q][dk]
  __bf16* Kb  = (__bf16*)alloc(actB);   // [bh][k][dk]
  __bf16* Vt  = (__bf16*)alloc(actB);   // [bh][dk][k]  (transposed)
  __bf16* ctx = (__bf16*)alloc(actB);   // [b][q][d_model]

  auto cvt = [&](const float* in, __bf16* out, size_t n) {
    int blocks = (int)((n / 4 + 255) / 256);
    f32_to_bf16_kernel<<<blocks, 256, 0, stream>>>(in, out, n);
  };
  cvt(queries, Xq, (size_t)MROWS * DMODEL);
  cvt(keys,    Xk, (size_t)MROWS * DMODEL);
  cvt(values,  Xv, (size_t)MROWS * DMODEL);
  cvt(Wq, Wqb, (size_t)DMODEL * DMODEL);
  cvt(Wk, Wkb, (size_t)DMODEL * DMODEL);
  cvt(Wv, Wvb, (size_t)DMODEL * DMODEL);
  cvt(Wo, Wob, (size_t)DMODEL * DMODEL);

  dim3 ggrid(MROWS / 256, DMODEL / 64, 1);   // 16 x 16
  gemm_nt_bf16<<<ggrid, 256, 0, stream>>>(Xq, Wqb, bq, qmask, Qb, nullptr, 0);
  gemm_nt_bf16<<<ggrid, 256, 0, stream>>>(Xk, Wkb, bk, kmask, Kb, nullptr, 1);
  gemm_nt_bf16<<<ggrid, 256, 0, stream>>>(Xv, Wvb, bv, kmask, Vt, nullptr, 2);

  attn_scores_kernel<<<dim3(NSEQ / 512, NSEQ / 16, B_SZ * NHEAD), 256, 0, stream>>>(
      Qb, Kb, attw, amask, att);

  softmax_rows_kernel<<<B_SZ * NHEAD * NSEQ, 256, 0, stream>>>(att);

  attn_pv_kernel<<<dim3(NSEQ / 256, B_SZ * NHEAD), 256, 0, stream>>>(att, Vt, ctx);

  gemm_nt_bf16<<<ggrid, 256, 0, stream>>>(ctx, Wob, bo, nullptr, nullptr, outF, 3);
}
